// model_0_3453153706228
// MI455X (gfx1250) — matrix operand order repaired and checked by reading, not compiled, not run on hardware
//
#include <hip/hip_runtime.h>
#include <cstdint>
#include <cstddef>

// ---------------- model constants ----------------
constexpr int B  = 4;
constexpr int S  = 1024;
constexpr int D  = 4096;
constexpr int H  = 32;
constexpr int KV = 8;
constexpr int HD = 128;
constexpr int FF = 14336;
constexpr int M  = B * S;
constexpr float EPS = 1e-5f;

// ---------------- types ----------------
typedef __attribute__((ext_vector_type(16))) __bf16 v16bf;
typedef __attribute__((ext_vector_type(8)))  float  v8f;
typedef int v4i __attribute__((vector_size(16)));

union BF16x16 { uint4 u[2]; __bf16 e[16]; v16bf v; };

static __device__ __forceinline__ __bf16 f2bf(float f) {
  union { float f; uint32_t u; } x{f};
  uint32_t r = x.u + 0x7FFFu + ((x.u >> 16) & 1u);   // round-to-nearest-even
  uint16_t h = (uint16_t)(r >> 16);
  __bf16 o; __builtin_memcpy(&o, &h, 2); return o;
}

static __device__ __forceinline__ v8f v8zero() {
  v8f v;
  #pragma unroll
  for (int i = 0; i < 8; ++i) v[i] = 0.f;
  return v;
}

// ---------------- CDNA5 async global->LDS copy (ASYNCcnt path) ----------------
#define AS1 __attribute__((address_space(1)))
#define AS3 __attribute__((address_space(3)))

static __device__ __forceinline__ void async_cp16(const void* g, void* l) {
#if __has_builtin(__builtin_amdgcn_global_load_async_to_lds_b128)
  // builtin expects int4* pointers (global src, LDS dst); go through uintptr_t:
  // AS1 pointers are 64-bit (value preserved), AS3 pointers are 32-bit (the
  // truncation of a generic shared pointer is exactly the LDS byte offset).
  __builtin_amdgcn_global_load_async_to_lds_b128(
      (AS1 v4i*)(uintptr_t)g, (AS3 v4i*)(uintptr_t)l, 0, 0);
#else
  uint32_t la = (uint32_t)(uintptr_t)l;
  uint64_t ga = (uint64_t)(uintptr_t)g;
  asm volatile("global_load_async_to_lds_b128 %0, %1, off"
               :: "v"(la), "v"(ga) : "memory");
#endif
}

static __device__ __forceinline__ void wait_async_le12() {
#if __has_builtin(__builtin_amdgcn_s_wait_asynccnt)
  __builtin_amdgcn_s_wait_asynccnt(12);
#else
  asm volatile("s_wait_asynccnt 0xc" ::: "memory");
#endif
}
static __device__ __forceinline__ void wait_async_le0() {
#if __has_builtin(__builtin_amdgcn_s_wait_asynccnt)
  __builtin_amdgcn_s_wait_asynccnt(0);
#else
  asm volatile("s_wait_asynccnt 0x0" ::: "memory");
#endif
}

// ---------------- DPP butterfly reductions within 16-lane rows ----------------
// quad_perm xor1 (0xB1), xor2 (0x4E), then row_ror:4 (0x124), row_ror:8 (0x128)
static __device__ __forceinline__ float rowmax16(float v) {
  int t;
  t = __builtin_amdgcn_update_dpp(0, __float_as_int(v), 0xB1, 0xF, 0xF, true);
  v = fmaxf(v, __int_as_float(t));
  t = __builtin_amdgcn_update_dpp(0, __float_as_int(v), 0x4E, 0xF, 0xF, true);
  v = fmaxf(v, __int_as_float(t));
  t = __builtin_amdgcn_update_dpp(0, __float_as_int(v), 0x124, 0xF, 0xF, true);
  v = fmaxf(v, __int_as_float(t));
  t = __builtin_amdgcn_update_dpp(0, __float_as_int(v), 0x128, 0xF, 0xF, true);
  v = fmaxf(v, __int_as_float(t));
  return v;
}
static __device__ __forceinline__ float rowsum16(float v) {
  int t;
  t = __builtin_amdgcn_update_dpp(0, __float_as_int(v), 0xB1, 0xF, 0xF, true);
  v = v + __int_as_float(t);
  t = __builtin_amdgcn_update_dpp(0, __float_as_int(v), 0x4E, 0xF, 0xF, true);
  v = v + __int_as_float(t);
  t = __builtin_amdgcn_update_dpp(0, __float_as_int(v), 0x124, 0xF, 0xF, true);
  v = v + __int_as_float(t);
  t = __builtin_amdgcn_update_dpp(0, __float_as_int(v), 0x128, 0xF, 0xF, true);
  v = v + __int_as_float(t);
  return v;
}

// ---------------- embedding gather ----------------
__global__ __launch_bounds__(256) void embed_k(const int* __restrict__ tok,
                                               const float* __restrict__ emb,
                                               float* __restrict__ h) {
  int row = blockIdx.x;
  int t = tok[row];
  const float4* src = (const float4*)(emb + (size_t)t * D);
  float4* dst = (float4*)(h + (size_t)row * D);
  for (int i = threadIdx.x; i < D / 4; i += 256) dst[i] = src[i];
}

// ---------------- fp32 -> bf16 elementwise ----------------
__global__ __launch_bounds__(256) void cvt_bf16_k(const float* __restrict__ src,
                                                  __bf16* __restrict__ dst, size_t n) {
  size_t i = ((size_t)blockIdx.x * 256 + threadIdx.x) * 4;
  if (i >= n) return;
  float4 f = *(const float4*)(src + i);
  dst[i + 0] = f2bf(f.x); dst[i + 1] = f2bf(f.y);
  dst[i + 2] = f2bf(f.z); dst[i + 3] = f2bf(f.w);
}

// ---------------- rmsnorm: fp32 row -> bf16 row ----------------
__global__ __launch_bounds__(256) void rmsnorm_k(const float* __restrict__ x,
                                                 const float* __restrict__ w,
                                                 __bf16* __restrict__ o) {
  __shared__ float red[256];
  int row = blockIdx.x;
  const float* xr = x + (size_t)row * D;
  float s = 0.f;
  for (int i = threadIdx.x; i < D; i += 256) { float v = xr[i]; s += v * v; }
  red[threadIdx.x] = s; __syncthreads();
  for (int off = 128; off > 0; off >>= 1) {
    if (threadIdx.x < off) red[threadIdx.x] += red[threadIdx.x + off];
    __syncthreads();
  }
  float inv = rsqrtf(red[0] / (float)D + EPS);
  for (int i = threadIdx.x; i < D; i += 256)
    o[(size_t)row * D + i] = f2bf(xr[i] * inv * w[i]);
}

// ---------------- RoPE (in place, fp32) ----------------
__global__ __launch_bounds__(256) void rope_k(float* __restrict__ x,
                                              const int* __restrict__ startp, int nh) {
  int idx = blockIdx.x * 256 + threadIdx.x;
  int i2 = idx & 63;               // HD/2 = 64
  int t  = idx >> 6;               // (b*S+s)*nh + head
  int bs = t / nh;
  int s  = bs & (S - 1);
  float pos  = (float)(startp[0] + s);
  float inv  = exp2f((-(2.f * (float)i2) / (float)HD) * 13.287712379549449f); // log2(1e4)
  float ang  = pos * inv;
  float sn, c; __sincosf(ang, &sn, &c);
  size_t base = (size_t)t * HD + 2 * i2;
  float x0 = x[base], x1 = x[base + 1];
  x[base]     = x0 * c - x1 * sn;
  x[base + 1] = x0 * sn + x1 * c;
}

// ---------------- WMMA bf16 GEMM: C[M,N] = A[M,K] * W[N,K]^T ----------------
// 256x128 block tile, 64x64 per-wave tile (4x4 fragments -> 1 ds_load per wmma),
// double-buffered LDS staged by async global->LDS copies (ASYNCcnt), BK=64.
// MODE 0: out fp32 = acc
// MODE 1: out fp32 = extra + acc          (residual)
// MODE 2: out bf16 = silu(extra) * acc    (fused SwiGLU epilogue)
constexpr int BK = 64;
constexpr int BM = 256;
constexpr int BN = 128;

template<int MODE>
__global__ __launch_bounds__(256) void gemm_k(const __bf16* __restrict__ A,
                                              const __bf16* __restrict__ Wt,
                                              void* __restrict__ outp,
                                              const float* __restrict__ extra,
                                              int Ndim, int Kdim) {
  __shared__ __attribute__((aligned(16))) __bf16 As[2][BM * BK];
  __shared__ __attribute__((aligned(16))) __bf16 Bs[2][BN * BK];

  const int bn = blockIdx.x * BN, bm = blockIdx.y * BM;
  const int tid  = threadIdx.x;
  const int wave = tid >> 5, lane = tid & 31;
  const int wm = (wave & 3) * 64;   // 4 waves along M -> 64 rows each
  const int wn = (wave >> 2) * 64;  // 2 waves along N -> 64 cols each
  const int mrow = lane & 15, half = lane >> 4;
  const int c0   = half * 8;        // ISA-table K base (unused; A-frag uses kb16)
  const int kb16 = half * 16;       // A/B-frag K base, contiguous k = 16*half + s

  v8f acc[4][4];
  #pragma unroll
  for (int a = 0; a < 4; ++a)
    #pragma unroll
    for (int b = 0; b < 4; ++b) acc[a][b] = v8zero();

  // staging: A: one 64-col row per thread (8 async b128); B: half-row (4 ops)
  const int arow = tid;               // 0..255
  const int brow = tid >> 1;          // 0..127
  const int bcol = (tid & 1) * 32;    // 0 or 32 bf16 elements
  const __bf16* aptr = A  + (size_t)(bm + arow) * Kdim;
  const __bf16* bptr = Wt + (size_t)(bn + brow) * Kdim + bcol;
  __bf16* asl = &As[0][arow * BK];
  __bf16* bsl = &Bs[0][brow * BK + bcol];

  auto stage = [&](int k0, int buf) {
    const __bf16* ga = aptr + k0;
    const __bf16* gb = bptr + k0;
    __bf16* la = asl + buf * (BM * BK);
    __bf16* lb = bsl + buf * (BN * BK);
    #pragma unroll
    for (int s8 = 0; s8 < BK; s8 += 8) async_cp16(ga + s8, la + s8);
    #pragma unroll
    for (int s8 = 0; s8 < 32; s8 += 8) async_cp16(gb + s8, lb + s8);
  };

  stage(0, 0);
  const int nIter = Kdim / BK;
  for (int it = 0; it < nIter; ++it) {
    const int cur = it & 1;
    if (it + 1 < nIter) { stage((it + 1) * BK, cur ^ 1); wait_async_le12(); }
    else                { wait_async_le0(); }
    __syncthreads();

    #pragma unroll
    for (int ks = 0; ks < BK; ks += 32) {
      v16bf af[4];
      #pragma unroll
      for (int mt = 0; mt < 4; ++mt) {
        BF16x16 t;
        const __bf16* p = &As[cur][(wm + mt * 16 + mrow) * BK + ks];
        t.u[0] = *(const uint4*)(p + kb16);
        t.u[1] = *(const uint4*)(p + kb16 + 8);
        af[mt] = t.v;
      }
      #pragma unroll
      for (int nt = 0; nt < 4; ++nt) {
        BF16x16 bf;
        const __bf16* p = &Bs[cur][(wn + nt * 16 + mrow) * BK + ks + kb16];
        bf.u[0] = *(const uint4*)(p);
        bf.u[1] = *(const uint4*)(p + 8);
        #pragma unroll
        for (int mt = 0; mt < 4; ++mt)
          acc[mt][nt] = __builtin_amdgcn_wmma_f32_16x16x32_bf16(
              false, af[mt], false, bf.v, (short)0, acc[mt][nt], false, false);
      }
    }
    __syncthreads();
  }

  #pragma unroll
  for (int mt = 0; mt < 4; ++mt)
    #pragma unroll
    for (int nt = 0; nt < 4; ++nt)
      #pragma unroll
      for (int r = 0; r < 8; ++r) {
        int row = bm + wm + mt * 16 + r + 8 * half;
        int col = bn + wn + nt * 16 + mrow;
        size_t idx = (size_t)row * Ndim + col;
        float v = acc[mt][nt][r];
        if (MODE == 0) {
          ((float*)outp)[idx] = v;
        } else if (MODE == 1) {
          ((float*)outp)[idx] = extra[idx] + v;
        } else {
          float u = extra[idx];
          float si = u * (1.f / (1.f + __expf(-u)));
          ((__bf16*)outp)[idx] = f2bf(si * v);
        }
      }
}

// ---------------- flash attention (one wave per b,head,16-row Q block) ----------------
__global__ __launch_bounds__(32) void attn_k(const float* __restrict__ q,
                                             const float* __restrict__ k,
                                             const float* __restrict__ v,
                                             float* __restrict__ o) {
  __shared__ __attribute__((aligned(16))) __bf16 Ks[32 * HD];  // [key][hd]
  __shared__ __attribute__((aligned(16))) __bf16 Vt[HD * 32];  // [hd][key] (transposed)
  __shared__ __attribute__((aligned(16))) __bf16 Ps[16 * 32];

  const int qb = blockIdx.x, hh = blockIdx.y, b = blockIdx.z;
  const int g = hh / (H / KV);
  const int lane = threadIdx.x;
  const int mrow = lane & 15, half = lane >> 4;
  const int c0 = half * 8, kb16 = half * 16;
  const float scale = 0.08838834764831845f;   // 1/sqrt(128)

  // Q tile 16x128 as 4 bf16 A-fragments (pre-scaled)
  v16bf qf[4];
  {
    const float* qrow = q + (((size_t)(b * S + qb * 16 + mrow)) * H + hh) * HD;
    #pragma unroll
    for (int c = 0; c < 4; ++c) {
      BF16x16 f;
      #pragma unroll
      for (int j = 0; j < 8; ++j) f.e[j]     = f2bf(qrow[c * 32 + kb16 + j] * scale);
      #pragma unroll
      for (int j = 0; j < 8; ++j) f.e[8 + j] = f2bf(qrow[c * 32 + kb16 + 8 + j] * scale);
      qf[c] = f.v;
    }
  }

  v8f acc[8];
  #pragma unroll
  for (int t = 0; t < 8; ++t) acc[t] = v8zero();
  float mrw[8], lrw[8];
  #pragma unroll
  for (int r = 0; r < 8; ++r) { mrw[r] = -1e30f; lrw[r] = 0.f; }

  const int nkc = (qb + 2) >> 1;          // causal: chunks of 32 keys
  for (int kc = 0; kc < nkc; ++kc) {
    __syncthreads();                       // prior P@V reads done before restage
    {   // K: one key row per lane, packed b128 LDS stores
      int key = kc * 32 + lane;
      const float* krow = k + (((size_t)(b * S + key)) * KV + g) * HD;
      #pragma unroll
      for (int d = 0; d < HD; d += 8) {
        union { uint4 u; __bf16 e[8]; } pk;
        #pragma unroll
        for (int j = 0; j < 8; ++j) pk.e[j] = f2bf(krow[d + j]);
        *(uint4*)&Ks[lane * HD + d] = pk.u;
      }
    }
    {   // V transposed: lane owns hd-columns d = j*32+lane; coalesced global reads,
        // packed b128 LDS stores, so PV B-frags become 2x ds_load_b128
      const float* vbase = v + (((size_t)(b * S + kc * 32)) * KV + g) * HD;
      #pragma unroll
      for (int j = 0; j < 4; ++j) {
        int d = j * 32 + lane;
        union { uint4 u[4]; __bf16 e[32]; } col;
        #pragma unroll
        for (int key = 0; key < 32; ++key)
          col.e[key] = f2bf(vbase[(size_t)key * KV * HD + d]);
        uint4* dst = (uint4*)&Vt[d * 32];
        dst[0] = col.u[0]; dst[1] = col.u[1]; dst[2] = col.u[2]; dst[3] = col.u[3];
      }
    }
    __syncthreads();

    // scores: two 16x16 tiles over a 32-key chunk, contracting HD=128 in 4 wmma
    v8f sc[2];
    #pragma unroll
    for (int t = 0; t < 2; ++t) {
      sc[t] = v8zero();
      #pragma unroll
      for (int c = 0; c < 4; ++c) {
        BF16x16 bf;
        const __bf16* p = &Ks[(t * 16 + mrow) * HD + c * 32 + kb16];
        bf.u[0] = *(const uint4*)(p);
        bf.u[1] = *(const uint4*)(p + 8);
        sc[t] = __builtin_amdgcn_wmma_f32_16x16x32_bf16(
            false, qf[c], false, bf.v, (short)0, sc[t], false, false);
      }
    }

    // online softmax; rows live in (vgpr r, lane-half); DPP butterfly, no DS ops
    int kbase = kc * 32;
    #pragma unroll
    for (int r = 0; r < 8; ++r) {
      int qrow = qb * 16 + r + 8 * half;
      float s0 = sc[0][r], s1 = sc[1][r];
      if (kbase + mrow      > qrow) s0 = -1e30f;
      if (kbase + 16 + mrow > qrow) s1 = -1e30f;
      float rm = rowmax16(fmaxf(s0, s1));
      float mnew = fmaxf(mrw[r], rm);
      float al = __expf(mrw[r] - mnew);
      float p0 = __expf(s0 - mnew), p1 = __expf(s1 - mnew);
      float rs = rowsum16(p0 + p1);
      lrw[r] = lrw[r] * al + rs;
      mrw[r] = mnew;
      #pragma unroll
      for (int t = 0; t < 8; ++t) acc[t][r] = acc[t][r] * al;
      Ps[(r + 8 * half) * 32 + mrow]      = f2bf(p0);
      Ps[(r + 8 * half) * 32 + 16 + mrow] = f2bf(p1);
    }
    __syncthreads();

    // P @ V : A=P (16x32), B=V chunk (32x16 per hd-tile), 8 wmma
    BF16x16 pa;
    pa.u[0] = *(const uint4*)(&Ps[mrow * 32 + kb16]);
    pa.u[1] = *(const uint4*)(&Ps[mrow * 32 + kb16 + 8]);
    #pragma unroll
    for (int ht = 0; ht < 8; ++ht) {
      BF16x16 vb;
      const __bf16* p = &Vt[(ht * 16 + mrow) * 32 + kb16];
      vb.u[0] = *(const uint4*)(p);
      vb.u[1] = *(const uint4*)(p + 8);
      acc[ht] = __builtin_amdgcn_wmma_f32_16x16x32_bf16(
          false, pa.v, false, vb.v, (short)0, acc[ht], false, false);
    }
  }

  #pragma unroll
  for (int r = 0; r < 8; ++r) {
    float inv = 1.f / lrw[r];
    int srow = qb * 16 + r + 8 * half;
    float* orow = o + (((size_t)(b * S + srow)) * H + hh) * HD;
    #pragma unroll
    for (int ht = 0; ht < 8; ++ht) orow[ht * 16 + mrow] = acc[ht][r] * inv;
  }
}

// ---------------- host launcher ----------------
extern "C" void kernel_launch(void* const* d_in, const int* in_sizes, int n_in,
                              void* d_out, int out_size, void* d_ws, size_t ws_size,
                              hipStream_t stream) {
  (void)in_sizes; (void)n_in; (void)out_size; (void)ws_size;
  const int*   tokens = (const int*)d_in[0];
  const int*   startp = (const int*)d_in[1];
  const float* emb_w  = (const float*)d_in[2];
  const float* wq     = (const float*)d_in[3];
  const float* wk     = (const float*)d_in[4];
  const float* wv     = (const float*)d_in[5];
  const float* wo     = (const float*)d_in[6];
  const float* w1     = (const float*)d_in[7];
  const float* w2     = (const float*)d_in[8];
  const float* w3     = (const float*)d_in[9];
  const float* anw    = (const float*)d_in[10];
  const float* fnw    = (const float*)d_in[11];

  char* ws = (char*)d_ws;
  size_t off = 0;
  auto alloc = [&](size_t bytes) { size_t o = off; off += (bytes + 255) & ~(size_t)255; return o; };

  __bf16* wq_bf = (__bf16*)(ws + alloc((size_t)D * D * 2));
  __bf16* wk_bf = (__bf16*)(ws + alloc((size_t)KV * HD * D * 2));
  __bf16* wv_bf = (__bf16*)(ws + alloc((size_t)KV * HD * D * 2));
  __bf16* wo_bf = (__bf16*)(ws + alloc((size_t)D * D * 2));
  __bf16* w1_bf = (__bf16*)(ws + alloc((size_t)FF * D * 2));
  __bf16* w2_bf = (__bf16*)(ws + alloc((size_t)D * FF * 2));
  __bf16* w3_bf = (__bf16*)(ws + alloc((size_t)FF * D * 2));
  float*  h     = (float*) (ws + alloc((size_t)M * D * 4));
  __bf16* xb    = (__bf16*)(ws + alloc((size_t)M * D * 2));
  float*  qb    = (float*) (ws + alloc((size_t)M * D * 4));
  float*  kb    = (float*) (ws + alloc((size_t)M * KV * HD * 4));
  float*  vb    = (float*) (ws + alloc((size_t)M * KV * HD * 4));
  float*  attn  = (float*) (ws + alloc((size_t)M * D * 4));
  __bf16* at_bf = (__bf16*)(ws + alloc((size_t)M * D * 2));
  float*  h2    = (float*) (ws + alloc((size_t)M * D * 4));
  __bf16* ybf   = (__bf16*)(ws + alloc((size_t)M * D * 2));
  float*  u1    = (float*) (ws + alloc((size_t)M * FF * 4));
  __bf16* zbf   = (__bf16*)(ws + alloc((size_t)M * FF * 2));

  auto cvt = [&](const float* s, __bf16* d, size_t n) {
    cvt_bf16_k<<<(unsigned)((n + 1023) / 1024), 256, 0, stream>>>(s, d, n);
  };

  // weight conversions (deterministic, redone each call)
  cvt(wq, wq_bf, (size_t)D * D);
  cvt(wk, wk_bf, (size_t)KV * HD * D);
  cvt(wv, wv_bf, (size_t)KV * HD * D);
  cvt(wo, wo_bf, (size_t)D * D);
  cvt(w1, w1_bf, (size_t)FF * D);
  cvt(w2, w2_bf, (size_t)D * FF);
  cvt(w3, w3_bf, (size_t)FF * D);

  embed_k<<<M, 256, 0, stream>>>(tokens, emb_w, h);
  rmsnorm_k<<<M, 256, 0, stream>>>(h, anw, xb);

  // QKV projections
  gemm_k<0><<<dim3(D / BN, M / BM), 256, 0, stream>>>(xb, wq_bf, qb, nullptr, D, D);
  gemm_k<0><<<dim3((KV * HD) / BN, M / BM), 256, 0, stream>>>(xb, wk_bf, kb, nullptr, KV * HD, D);
  gemm_k<0><<<dim3((KV * HD) / BN, M / BM), 256, 0, stream>>>(xb, wv_bf, vb, nullptr, KV * HD, D);

  // RoPE on q (H heads) and k (KV heads)
  rope_k<<<(M * H  * (HD / 2)) / 256, 256, 0, stream>>>(qb, startp, H);
  rope_k<<<(M * KV * (HD / 2)) / 256, 256, 0, stream>>>(kb, startp, KV);

  // flash attention
  attn_k<<<dim3(S / 16, H, B), 32, 0, stream>>>(qb, kb, vb, attn);

  // output projection + residual
  cvt(attn, at_bf, (size_t)M * D);
  gemm_k<1><<<dim3(D / BN, M / BM), 256, 0, stream>>>(at_bf, wo_bf, h2, h, D, D);

  // FFN
  rmsnorm_k<<<M, 256, 0, stream>>>(h2, fnw, ybf);
  gemm_k<0><<<dim3(FF / BN, M / BM), 256, 0, stream>>>(ybf, w1_bf, u1, nullptr, FF, D);
  gemm_k<2><<<dim3(FF / BN, M / BM), 256, 0, stream>>>(ybf, w3_bf, zbf, u1, FF, D);
  gemm_k<1><<<dim3(D / BN, M / BM), 256, 0, stream>>>(zbf, w2_bf, (float*)d_out, h2, D, FF);
}
